// SpeechVideoCrossAttention_79207786872842
// MI455X (gfx1250) — matrix operand order repaired and checked by reading, not compiled, not run on hardware
//
#include <hip/hip_runtime.h>
#include <hip/hip_bf16.h>

// ---------------- problem constants ----------------
#define BATCH   8192
#define EMB     2048
#define NHEAD   16
#define HDIM    128
#define NVID    10
#define SCALE   0.08838834764831845f   // 1/sqrt(128)

typedef __attribute__((ext_vector_type(16))) __bf16 v16bf;
typedef __attribute__((ext_vector_type(8)))  __bf16 v8bf;
typedef __attribute__((ext_vector_type(8)))  float  v8f;

union bfrag { v16bf v; v8bf h[2]; };

static __device__ __forceinline__ v8bf ld8(const __hip_bfloat16* p) {
    return *(const v8bf*)(const void*)p;
}

static __device__ __forceinline__ v8f wmma_bf16(const bfrag& a, const bfrag& b, v8f c) {
    return __builtin_amdgcn_wmma_f32_16x16x32_bf16(
        false, a.v, false, b.v, (short)0, c, false, false);
}

// ---- CDNA5 async global->LDS copy:
//      __builtin_amdgcn_global_load_async_to_lds_b128(int4 AS1* src, int4 AS3* dst, imm off, imm cpol)
#if defined(__AMDGCN__) && __has_builtin(__builtin_amdgcn_global_load_async_to_lds_b128) && \
    __has_builtin(__builtin_amdgcn_s_wait_asynccnt)
#define HAVE_ASYNC_LDS 1
typedef int v4i_g __attribute__((vector_size(16)));
typedef __attribute__((address_space(1))) v4i_g* gp128_t;
typedef __attribute__((address_space(3))) v4i_g* lp128_t;
static __device__ __forceinline__ gp128_t gp128(const void* p) {
    return (gp128_t)(unsigned long long)p;
}
static __device__ __forceinline__ lp128_t lp128(void* p) {
    // generic -> LDS: low 32 bits of the generic address are the LDS offset
    return (lp128_t)(unsigned int)(unsigned long long)p;
}
#else
#define HAVE_ASYNC_LDS 0
#endif

// ---------------- kernel 1: layernorm + cast to bf16 ----------------
__global__ void __launch_bounds__(256)
ln_cast_kernel(const float* __restrict__ src, const float* __restrict__ g,
               const float* __restrict__ beta, __hip_bfloat16* __restrict__ dst) {
    const size_t row = blockIdx.x;
    const float* x = src + row * EMB;
    const int t = threadIdx.x;

    float v[8];
    float s = 0.f, ss = 0.f;
#pragma unroll
    for (int i = 0; i < 8; ++i) {
        v[i] = x[t + 256 * i];
        s  += v[i];
        ss += v[i] * v[i];
    }
#pragma unroll
    for (int off = 16; off; off >>= 1) {
        s  += __shfl_xor(s,  off);
        ss += __shfl_xor(ss, off);
    }
    __shared__ float rs[8], rss[8];
    const int w = t >> 5, l = t & 31;
    if (l == 0) { rs[w] = s; rss[w] = ss; }
    __syncthreads();
    s = 0.f; ss = 0.f;
#pragma unroll
    for (int i = 0; i < 8; ++i) { s += rs[i]; ss += rss[i]; }

    const float mu  = s * (1.f / EMB);
    const float var = ss * (1.f / EMB) - mu * mu;
    const float inv = rsqrtf(var + 1e-5f);
#pragma unroll
    for (int i = 0; i < 8; ++i) {
        const int c = t + 256 * i;
        const float y = (v[i] - mu) * inv * g[c] + beta[c];
        dst[row * EMB + c] = __float2bfloat16(y);
    }
}

// ---------------- kernel 2: weight transpose + cast (W[k][n] -> WT[n][k] bf16) --
__global__ void __launch_bounds__(256)
transpose_cast_kernel(const float* __restrict__ W, __hip_bfloat16* __restrict__ WT) {
    __shared__ float tile[32][33];
    const int tx = threadIdx.x, ty = threadIdx.y;
    const int kb = blockIdx.y * 32, nb = blockIdx.x * 32;
#pragma unroll
    for (int i = 0; i < 32; i += 8)
        tile[ty + i][tx] = W[(size_t)(kb + ty + i) * EMB + nb + tx];
    __syncthreads();
#pragma unroll
    for (int i = 0; i < 32; i += 8)
        WT[(size_t)(nb + ty + i) * EMB + kb + tx] = __float2bfloat16(tile[tx][ty + i]);
}

// ---------------- kernel 3: bf16 WMMA GEMM ----------------
// Y[M,2048] = X[M,2048] @ W + bias, with WT = W^T pre-transposed ([N][K] bf16).
// grid.x = M/256 (8 waves x Mtile32), grid.y * (64*n_iters) = 2048.
// N loop inside the block so the block's 1MB X slice stays L2-resident.
// B tile (64x32 bf16, 4KB) double-buffered in LDS, staged with async-to-LDS.
// All 4 B fragments are held in distinct registers so the 8-wide ds clause
// drains while the 8 WMMAs issue (no per-subtile s_wait_dscnt 0 fences).
__global__ void __launch_bounds__(256)
gemm_bf16_kernel(const __hip_bfloat16* __restrict__ X,
                 const __hip_bfloat16* __restrict__ WT,
                 const float* __restrict__ bias,
                 float* __restrict__ Y, int n_iters) {
    __shared__ __align__(16) __hip_bfloat16 lds_b[2][64 * 32];

    const int t    = threadIdx.x;
    const int w    = t >> 5;
    const int l    = t & 31;
    const int half = l >> 4;          // 0 or 1
    const int lm   = l & 15;
    const int m0   = blockIdx.x * 256 + w * 32;

    // A fragment pointers (per ISA 16-bit A layout: lanes<16 K0-7/K16-23, lanes>=16 +8)
    const __hip_bfloat16* xrow0 = X + (size_t)(m0 + lm) * EMB + half * 8;
    const __hip_bfloat16* xrow1 = xrow0 + (size_t)16 * EMB;

    // cooperative staging: thread t copies 16B of WT row (n0 + t/4), chunk (t&3)
    const int sn = t >> 2;            // 0..63
    const int sc = (t & 3) * 8;       // element offset 0,8,16,24
    const int nbase = blockIdx.y * 64 * n_iters;

    for (int ni = 0; ni < n_iters; ++ni) {
        const int n0 = nbase + ni * 64;
        const __hip_bfloat16* wrow = WT + (size_t)(n0 + sn) * EMB + sc;

        v8f acc[8] = {};

        // prologue: stage k-tile 0 into buffer 0
#if HAVE_ASYNC_LDS
        __builtin_amdgcn_global_load_async_to_lds_b128(
            gp128(wrow), lp128(&lds_b[0][sn * 32 + sc]), 0, 0);
        __builtin_amdgcn_s_wait_asynccnt(0);
#else
        *(v8bf*)(void*)&lds_b[0][sn * 32 + sc] = ld8(wrow);
#endif
        __syncthreads();

        int cur = 0;
        for (int kk = 0; kk < EMB; kk += 32) {
            const bool more = (kk + 32) < EMB;
#if HAVE_ASYNC_LDS
            if (more)
                __builtin_amdgcn_global_load_async_to_lds_b128(
                    gp128(wrow + kk + 32), lp128(&lds_b[cur ^ 1][sn * 32 + sc]), 0, 0);
#else
            v8bf nxt;
            if (more) nxt = ld8(wrow + kk + 32);
#endif
            __builtin_prefetch((const void*)(xrow0 + kk + 256), 0, 3);

            bfrag A0, A1;
            A0.h[0] = ld8(xrow0 + kk);
            A0.h[1] = ld8(xrow0 + kk + 16);
            A1.h[0] = ld8(xrow1 + kk);
            A1.h[1] = ld8(xrow1 + kk + 16);

            // pull all 4 B fragments first (distinct regs -> one ds clause)
            bfrag Bf[4];
#pragma unroll
            for (int a = 0; a < 4; ++a) {
                const __hip_bfloat16* bp = &lds_b[cur][(a * 16 + lm) * 32 + half * 8];
                Bf[a].h[0] = *(const v8bf*)(const void*)bp;
                Bf[a].h[1] = *(const v8bf*)(const void*)(bp + 16);
            }
#pragma unroll
            for (int a = 0; a < 4; ++a) {
                acc[a]     = wmma_bf16(A0, Bf[a], acc[a]);
                acc[4 + a] = wmma_bf16(A1, Bf[a], acc[4 + a]);
            }

#if HAVE_ASYNC_LDS
            __builtin_amdgcn_s_wait_asynccnt(0);
#else
            if (more) *(v8bf*)(void*)&lds_b[cur ^ 1][sn * 32 + sc] = nxt;
#endif
            __syncthreads();
            cur ^= 1;
        }

        // C/D layout: VGPR i, lanes 0-15 -> M=i, lanes 16-31 -> M=i+8; N = lane%16
#pragma unroll
        for (int a = 0; a < 4; ++a) {
            const int col = n0 + a * 16 + lm;
            const float bv = bias[col];
            float* yp0 = Y + (size_t)(m0 + half * 8) * EMB + col;
            float* yp1 = yp0 + (size_t)16 * EMB;
#pragma unroll
            for (int i = 0; i < 8; ++i) {
                yp0[(size_t)i * EMB] = acc[a][i] + bv;
                yp1[(size_t)i * EMB] = acc[4 + a][i] + bv;
            }
        }
        __syncthreads();
    }
}

// ---------------- kernel 4: per-(batch,head) attention ----------------
__global__ void __launch_bounds__(256)
attn_kernel(const float* __restrict__ Qf, const float* __restrict__ Kf,
            const float* __restrict__ Vf, const float* __restrict__ alpha_p,
            __hip_bfloat16* __restrict__ ctxb, float* __restrict__ probs) {
    const int t = threadIdx.x;
    const int w = t >> 5, l = t & 31;
    const int gw = blockIdx.x * 8 + w;
    const int b = gw >> 4, h = gw & 15;
    const float alpha = alpha_p[0];

    const size_t qoff = (size_t)b * EMB + h * HDIM + l * 4;
    const float4 q = *(const float4*)(Qf + qoff);

    float sc[NVID];
#pragma unroll
    for (int k = 0; k < NVID; ++k) {
        const float4 kv = *(const float4*)(Kf + ((size_t)b * NVID + k) * EMB + h * HDIM + l * 4);
        float s = q.x * kv.x + q.y * kv.y + q.z * kv.z + q.w * kv.w;
#pragma unroll
        for (int off = 16; off; off >>= 1) s += __shfl_xor(s, off);
        float pb = fmaxf(-alpha * (float)(k * k), -10.f);
        sc[k] = fminf(fmaxf(s * SCALE, -1000.f), 1000.f) + pb;
    }

    float m = sc[0];
#pragma unroll
    for (int k = 1; k < NVID; ++k) m = fmaxf(m, sc[k]);
    float sum = 0.f;
#pragma unroll
    for (int k = 0; k < NVID; ++k) { sc[k] = __expf(sc[k] - m); sum += sc[k]; }
    const float inv = 1.f / sum;

    float4 c = make_float4(0.f, 0.f, 0.f, 0.f);
    float myp = 0.f;
#pragma unroll
    for (int k = 0; k < NVID; ++k) {
        const float p = sc[k] * inv;
        const float4 vv = *(const float4*)(Vf + ((size_t)b * NVID + k) * EMB + h * HDIM + l * 4);
        c.x += p * vv.x; c.y += p * vv.y; c.z += p * vv.z; c.w += p * vv.w;
        if (l == k) myp = p;
    }

    const size_t cbase = (size_t)b * EMB + h * HDIM + l * 4;
    ctxb[cbase + 0] = __float2bfloat16(c.x);
    ctxb[cbase + 1] = __float2bfloat16(c.y);
    ctxb[cbase + 2] = __float2bfloat16(c.z);
    ctxb[cbase + 3] = __float2bfloat16(c.w);
    if (l < NVID) probs[(size_t)(b * NHEAD + h) * NVID + l] = myp;
}

// ---------------- host-side launcher ----------------
extern "C" void kernel_launch(void* const* d_in, const int* in_sizes, int n_in,
                              void* d_out, int out_size, void* d_ws, size_t ws_size,
                              hipStream_t stream) {
    (void)in_sizes; (void)n_in; (void)out_size; (void)ws_size;

    const float* speech = (const float*)d_in[0];
    const float* video  = (const float*)d_in[1];
    const float* ln_s_g = (const float*)d_in[2];
    const float* ln_s_b = (const float*)d_in[3];
    const float* ln_v_g = (const float*)d_in[4];
    const float* ln_v_b = (const float*)d_in[5];
    const float* Wq = (const float*)d_in[6];
    const float* bq = (const float*)d_in[7];
    const float* Wk = (const float*)d_in[8];
    const float* bk = (const float*)d_in[9];
    const float* Wv = (const float*)d_in[10];
    const float* bv = (const float*)d_in[11];
    const float* Wo = (const float*)d_in[12];
    const float* bo = (const float*)d_in[13];
    const float* alpha = (const float*)d_in[14];

    float* out = (float*)d_out;

    char* ws = (char*)d_ws;
    size_t off = 0;
    auto take = [&](size_t bytes) { char* p = ws + off; off += (bytes + 255) & ~(size_t)255; return p; };

    __hip_bfloat16* spb  = (__hip_bfloat16*)take((size_t)BATCH * EMB * 2);
    __hip_bfloat16* vtb  = (__hip_bfloat16*)take((size_t)BATCH * NVID * EMB * 2);
    __hip_bfloat16* WqT  = (__hip_bfloat16*)take((size_t)EMB * EMB * 2);
    __hip_bfloat16* WkT  = (__hip_bfloat16*)take((size_t)EMB * EMB * 2);
    __hip_bfloat16* WvT  = (__hip_bfloat16*)take((size_t)EMB * EMB * 2);
    __hip_bfloat16* WoT  = (__hip_bfloat16*)take((size_t)EMB * EMB * 2);
    float*          Qf   = (float*)take((size_t)BATCH * EMB * 4);
    float*          Kf   = (float*)take((size_t)BATCH * NVID * EMB * 4);
    float*          Vf   = (float*)take((size_t)BATCH * NVID * EMB * 4);
    __hip_bfloat16* ctxb = (__hip_bfloat16*)take((size_t)BATCH * EMB * 2);

    // 1) layernorm + cast
    ln_cast_kernel<<<BATCH, 256, 0, stream>>>(speech, ln_s_g, ln_s_b, spb);
    ln_cast_kernel<<<BATCH * NVID, 256, 0, stream>>>(video, ln_v_g, ln_v_b, vtb);

    // 2) weight transpose + cast
    dim3 tgrid(EMB / 32, EMB / 32), tblk(32, 8);
    transpose_cast_kernel<<<tgrid, tblk, 0, stream>>>(Wq, WqT);
    transpose_cast_kernel<<<tgrid, tblk, 0, stream>>>(Wk, WkT);
    transpose_cast_kernel<<<tgrid, tblk, 0, stream>>>(Wv, WvT);
    transpose_cast_kernel<<<tgrid, tblk, 0, stream>>>(Wo, WoT);

    // 3) projections (bias fused). Small-M GEMMs split N 8-ways for occupancy
    //    (256 blocks); big-M GEMMs keep full-N loop so X stays L2-resident.
    gemm_bf16_kernel<<<dim3(BATCH / 256, 8), 256, 0, stream>>>(spb, WqT, bq, Qf, 4);
    gemm_bf16_kernel<<<dim3(BATCH * NVID / 256, 1), 256, 0, stream>>>(vtb, WkT, bk, Kf, 32);
    gemm_bf16_kernel<<<dim3(BATCH * NVID / 256, 1), 256, 0, stream>>>(vtb, WvT, bv, Vf, 32);

    // 4) attention
    float* probs_out = out + (size_t)BATCH * EMB;
    attn_kernel<<<BATCH * NHEAD / 8, 256, 0, stream>>>(Qf, Kf, Vf, alpha, ctxb, probs_out);

    // 5) output projection into d_out
    gemm_bf16_kernel<<<dim3(BATCH / 256, 8), 256, 0, stream>>>(ctxb, WoT, bo, out, 4);
}
